// EgoLSTM_39006892982731
// MI455X (gfx1250) — matrix operand order repaired and checked by reading, not compiled, not run on hardware
//
#include <hip/hip_runtime.h>

// ---- Problem constants (from reference) ----
#define B_TOT   8192
#define T_ENC   50
#define D_INP   6
#define HD      128
#define D_OUT2  2
#define OUT_LEN 60
#define G4      512     // 4*H
#define BT      64      // batch rows per workgroup
#define NTHR    256     // 8 wave32 per workgroup

typedef _Float16 v16h __attribute__((ext_vector_type(16)));
typedef _Float16 v8h  __attribute__((ext_vector_type(8)));
typedef float    v8f  __attribute__((ext_vector_type(8)));

// LDS layout (bytes):
//  sWhh : G4*HD f16          = 131072
//  sWih : G4*32 f16 (padded) =  32768
//  sH   : 2 * BT*HD f16      =  32768   (double-buffered hidden state)
//  sBias: G4 f32             =   2048
//  sPW  : 2*HD f32           =   1024
//  sPB  : 2 f32 (+pad)       =     16
//  sPred: BT*2 f32           =    512
#define SMEM_BYTES 200704

__device__ __forceinline__ float sigm(float x) { return 1.0f / (1.0f + __expf(-x)); }

// A-fragment (16x32 f16, M x K) from row-major f16 hidden state in LDS.
// Lane L (L<16):  M = L,      K = {kf*32+0..15}
// Lane L (L>=16): M = L-16,   K = {kf*32+16..31}
__device__ __forceinline__ v16h load_afrag(const _Float16* __restrict__ hbuf,
                                           int lane, int mtile, int kfrag) {
    const int row = mtile * 16 + (lane & 15);
    const int kh  = kfrag * 32 + ((lane >> 4) << 4);
    const _Float16* p = hbuf + row * HD;
    v8h lo = *(const v8h*)(p + kh);
    v8h hi = *(const v8h*)(p + kh + 8);
    return __builtin_shufflevector(lo, hi, 0,1,2,3,4,5,6,7,8,9,10,11,12,13,14,15);
}

// B-fragment (32x16 f16, K x N) from row-major f16 weights (W[n][k], gates = h @ W^T).
// Lane L: N = ntile*16 + (L&15); K = (L>>4)*16 + {0..15}  -> 16 contiguous halves.
__device__ __forceinline__ v16h load_bfrag(const _Float16* __restrict__ W,
                                           int lane, int ntile, int kbase, int stride) {
    const int n  = ntile * 16 + (lane & 15);
    const int kb = kbase + ((lane >> 4) << 4);
    return *(const v16h*)(W + n * stride + kb);
}

__device__ __forceinline__ void load_phase_weights(const float* __restrict__ Wih,
                                                   const float* __restrict__ Whh,
                                                   const float* __restrict__ bih,
                                                   const float* __restrict__ bhh,
                                                   int din,
                                                   _Float16* __restrict__ sWih,
                                                   _Float16* __restrict__ sWhh,
                                                   float* __restrict__ sBias,
                                                   int tid) {
    for (int i = tid; i < G4 * HD; i += NTHR) sWhh[i] = (_Float16)Whh[i];
    for (int i = tid; i < G4 * 32; i += NTHR) {
        int n = i >> 5, k = i & 31;
        sWih[i] = (k < din) ? (_Float16)Wih[n * din + k] : (_Float16)0.0f;
    }
    for (int i = tid; i < G4; i += NTHR) sBias[i] = bih[i] + bhh[i];
}

// One LSTM cell step for this wave's (mtile, jbase..jbase+3) ownership.
// gates = [ax @ Wih_pad^T] + [h @ Whh^T] + b, done with 16x16x32 f16 WMMA.
__device__ __forceinline__ void lstm_cell_step(const _Float16* __restrict__ sWhh,
                                               const _Float16* __restrict__ sWih,
                                               const float* __restrict__ sBias,
                                               const _Float16* __restrict__ hR,
                                               _Float16* __restrict__ hW,
                                               v16h ax, v8f (&creg)[4],
                                               int lane, int mtile, int jbase) {
    const int l15 = lane & 15;
    v16h ah[4];
#pragma unroll
    for (int kf = 0; kf < 4; kf++) ah[kf] = load_afrag(hR, lane, mtile, kf);

#pragma unroll
    for (int jj = 0; jj < 4; jj++) {
        const int j = jbase + jj;
        v8f acc[4];
#pragma unroll
        for (int g = 0; g < 4; g++) {                 // i, f, g, o gate blocks
            const int ntile = j + g * 8;
            const float bv = sBias[ntile * 16 + l15]; // bias per output column
            v8f c;
#pragma unroll
            for (int p = 0; p < 8; p++) c[p] = bv;
            // input contribution (K padded to 32)
            v16h bx = load_bfrag(sWih, lane, ntile, 0, 32);
            c = __builtin_amdgcn_wmma_f32_16x16x32_f16(false, ax, false, bx,
                                                       (short)0, c, false, false);
            // recurrent contribution, K = 128 = 4 x 32
#pragma unroll
            for (int kf = 0; kf < 4; kf++) {
                v16h bw = load_bfrag(sWhh, lane, ntile, kf * 32, HD);
                c = __builtin_amdgcn_wmma_f32_16x16x32_f16(false, ah[kf], false, bw,
                                                           (short)0, c, false, false);
            }
            acc[g] = c;
        }
        // Elementwise on matching fragment slots: lane holds col (unit), 8 rows.
        const int unit  = j * 16 + l15;
        const int rbase = mtile * 16 + ((lane >> 4) << 3);
#pragma unroll
        for (int p = 0; p < 8; p++) {
            float iv = sigm(acc[0][p]);
            float fv = sigm(acc[1][p]);
            float gv = tanhf(acc[2][p]);
            float ov = sigm(acc[3][p]);
            float cn = fv * creg[jj][p] + iv * gv;
            creg[jj][p] = cn;
            hW[(rbase + p) * HD + unit] = (_Float16)(ov * tanhf(cn));
        }
    }
}

extern "C" __global__ void __launch_bounds__(NTHR)
ego_lstm_persistent(const float* __restrict__ x,
                    const float* __restrict__ eWih, const float* __restrict__ eWhh,
                    const float* __restrict__ ebih, const float* __restrict__ ebhh,
                    const float* __restrict__ dWih, const float* __restrict__ dWhh,
                    const float* __restrict__ dbih, const float* __restrict__ dbhh,
                    const float* __restrict__ pW,   const float* __restrict__ pb,
                    float* __restrict__ out) {
    extern __shared__ char smem[];
    _Float16* sWhh = (_Float16*)smem;            // 512*128 halves
    _Float16* sWih = sWhh + G4 * HD;             // 512*32 halves (K padded)
    _Float16* sH   = sWih + G4 * 32;             // 2 * 64*128 halves
    float* sBias = (float*)(sH + 2 * BT * HD);   // 512 f32
    float* sPW   = sBias + G4;                   // 2*128 f32
    float* sPB   = sPW + D_OUT2 * HD;            // 2 f32
    float* sPred = sPB + 4;                      // 64*2 f32

    const int tid   = threadIdx.x;
    const int lane  = tid & 31;
    const int w     = tid >> 5;
    const int mtile = w >> 1;
    const int jbase = (w & 1) * 4;
    const int l15   = lane & 15;
    const int rowBase = blockIdx.x * BT;

    // ---- stage encoder weights + projection weights, zero h ----
    load_phase_weights(eWih, eWhh, ebih, ebhh, D_INP, sWih, sWhh, sBias, tid);
    for (int i = tid; i < D_OUT2 * HD; i += NTHR) sPW[i] = pW[i];
    if (tid < 2) sPB[tid] = pb[tid];
    for (int i = tid; i < BT * HD; i += NTHR) sH[i] = (_Float16)0.0f;
    __syncthreads();

    v8f creg[4];
#pragma unroll
    for (int jj = 0; jj < 4; jj++)
#pragma unroll
        for (int p = 0; p < 8; p++) creg[jj][p] = 0.0f;

    int rb = 0;  // read-buffer index into double-buffered sH

    // ================= encoder: 50 steps =================
    for (int t = 0; t < T_ENC; t++) {
        const _Float16* hR = sH + rb * BT * HD;
        _Float16* hW = sH + (rb ^ 1) * BT * HD;

        v16h ax;
#pragma unroll
        for (int q = 0; q < 16; q++) ax[q] = (_Float16)0.0f;
        if (lane < 16) {  // lanes>=16 carry K=16..31 -> all zero (D_INP=6)
            const float* xp = x + ((size_t)(rowBase + mtile * 16 + l15) * T_ENC + t) * D_INP;
#pragma unroll
            for (int k = 0; k < D_INP; k++) ax[k] = (_Float16)xp[k];
        }
        lstm_cell_step(sWhh, sWih, sBias, hR, hW, ax, creg, lane, mtile, jbase);
        __syncthreads();
        rb ^= 1;
    }

    // ---- inp0 = x[:, -1, :2]; swap in decoder weights ----
    if (tid < BT * D_OUT2) {
        int r = tid >> 1, o = tid & 1;
        sPred[tid] = x[((size_t)(rowBase + r) * T_ENC + (T_ENC - 1)) * D_INP + o];
    }
    __syncthreads();
    load_phase_weights(dWih, dWhh, dbih, dbhh, D_OUT2, sWih, sWhh, sBias, tid);
    __syncthreads();

    // ================= decoder: 60 steps =================
    for (int s = 0; s < OUT_LEN; s++) {
        const _Float16* hR = sH + rb * BT * HD;
        _Float16* hW = sH + (rb ^ 1) * BT * HD;

        v16h ax;
#pragma unroll
        for (int q = 0; q < 16; q++) ax[q] = (_Float16)0.0f;
        if (lane < 16) {
            int r = mtile * 16 + l15;
            ax[0] = (_Float16)sPred[r * 2 + 0];
            ax[1] = (_Float16)sPred[r * 2 + 1];
        }
        lstm_cell_step(sWhh, sWih, sBias, hR, hW, ax, creg, lane, mtile, jbase);
        __syncthreads();

        // projection: pred = h_new @ projW^T + projB  (2 outputs per row)
        if (tid < BT * D_OUT2) {
            int r = tid >> 1, o = tid & 1;
            float acc = sPB[o];
            const _Float16* hp = hW + r * HD;
#pragma unroll 8
            for (int u = 0; u < HD; u++) acc += (float)hp[u] * sPW[o * HD + u];
            out[(((size_t)(rowBase + r)) * OUT_LEN + s) * D_OUT2 + o] = acc;
            sPred[tid] = acc;  // autoregressive feedback
        }
        __syncthreads();
        rb ^= 1;
    }
}

extern "C" void kernel_launch(void* const* d_in, const int* in_sizes, int n_in,
                              void* d_out, int out_size, void* d_ws, size_t ws_size,
                              hipStream_t stream) {
    (void)in_sizes; (void)n_in; (void)d_ws; (void)ws_size; (void)out_size;
    const float* x    = (const float*)d_in[0];
    const float* eWih = (const float*)d_in[1];
    const float* eWhh = (const float*)d_in[2];
    const float* ebih = (const float*)d_in[3];
    const float* ebhh = (const float*)d_in[4];
    const float* dWih = (const float*)d_in[5];
    const float* dWhh = (const float*)d_in[6];
    const float* dbih = (const float*)d_in[7];
    const float* dbhh = (const float*)d_in[8];
    const float* pW   = (const float*)d_in[9];
    const float* pb   = (const float*)d_in[10];
    float* out = (float*)d_out;

    // Opt in to large dynamic LDS (host-side attribute set; not a stream op).
    static bool attr_set = false;
    if (!attr_set) {
        (void)hipFuncSetAttribute((const void*)ego_lstm_persistent,
                                  hipFuncAttributeMaxDynamicSharedMemorySize,
                                  SMEM_BYTES);
        attr_set = true;
    }

    dim3 grid(B_TOT / BT);   // 128 workgroups
    dim3 block(NTHR);        // 8 wave32
    ego_lstm_persistent<<<grid, block, SMEM_BYTES, stream>>>(
        x, eWih, eWhh, ebih, ebhh, dWih, dWhh, dbih, dbhh, pW, pb, out);
}
